// MHA_88742614270713
// MI455X (gfx1250) — matrix operand order repaired and checked by reading, not compiled, not run on hardware
//
#include <hip/hip_runtime.h>
#include <hip/hip_bf16.h>

// ---- problem constants (match reference) ----
#define BB   2
#define SS   2048
#define DM   1024
#define HH   16
#define DH   64

// ---- WMMA / vector types (CDNA5 gfx1250, wave32) ----
typedef __attribute__((ext_vector_type(16))) __bf16         v16bf;
typedef __attribute__((ext_vector_type(16))) unsigned short v16u;
typedef __attribute__((ext_vector_type(8)))  unsigned short v8u;
typedef __attribute__((ext_vector_type(4)))  unsigned short v4u;
typedef __attribute__((ext_vector_type(8)))  float          v8f;
typedef __attribute__((ext_vector_type(2)))  float          v2f;

__device__ __forceinline__ unsigned short f2bf(float x) {
    __bf16 h = (__bf16)x;                      // native v_cvt on gfx1250
    return __builtin_bit_cast(unsigned short, h);
}

__device__ __forceinline__ v8f wmma_bf16(v16u a, v16u b, v8f c) {
    return __builtin_amdgcn_wmma_f32_16x16x32_bf16(
        /*neg_a=*/false, __builtin_bit_cast(v16bf, a),
        /*neg_b=*/false, __builtin_bit_cast(v16bf, b),
        /*c_mod=*/(short)0, c, /*reuse_a=*/false, /*reuse_b=*/false);
}

// A-matrix (16x32 bf16) per-lane K index for element-pair p (0..7), lane-half hi.
__device__ __forceinline__ int a_kidx(int p, int hi) {
    return hi * 16 + p * 2;
}

// 16 contiguous bf16 from LDS, 16B-aligned base -> 2x ds_load_b128
__device__ __forceinline__ v16u ld16_lds(const unsigned short* p) {
    v8u a = *(const v8u*)p;
    v8u b = *(const v8u*)(p + 8);
    v16u r;
    #pragma unroll
    for (int i = 0; i < 8; ++i) { r[i] = a[i]; r[i + 8] = b[i]; }
    return r;
}

// ---- CDNA5 async global->LDS (ISA 08_async_tensor: GV mode, ASYNCcnt) ----
// Each active lane copies 16B: global source to the given LDS offset.
__device__ __forceinline__ void async_copy_b128(const unsigned short* lds_dst,
                                                const void* gaddr) {
    unsigned lds_off = (unsigned)(unsigned long long)lds_dst; // addr[31:0] = LDS offset
    asm volatile("global_load_async_to_lds_b128 %0, %1, off"
                 :: "v"(lds_off), "v"(gaddr) : "memory");
}
__device__ __forceinline__ void wait_async0() {
    asm volatile("s_wait_asynccnt 0" ::: "memory");
}

// =====================================================================
// Kernel 1: X @ W + b for Q/K/V (blockIdx.z selects projection).
// 128x64 tile per block, 8 waves, K-step 32, bf16 WMMA, f32 accum.
// Qh/Kh -> [B,H,S,64] bf16 (coalesced via LDS stage); V -> Vt [B,H,64,S].
// =====================================================================
__global__ __launch_bounds__(256) void proj_kernel(
    const float* __restrict__ q, const float* __restrict__ k, const float* __restrict__ v,
    const float* __restrict__ wq, const float* __restrict__ bq,
    const float* __restrict__ wk, const float* __restrict__ bk,
    const float* __restrict__ wv, const float* __restrict__ bv,
    unsigned short* __restrict__ Qh, unsigned short* __restrict__ Kh,
    unsigned short* __restrict__ Vt)
{
    const int z = blockIdx.z;
    const float* X  = (z == 0) ? q  : (z == 1) ? k  : v;
    const float* W  = (z == 0) ? wq : (z == 1) ? wk : wv;
    const float* Bi = (z == 0) ? bq : (z == 1) ? bk : bv;

    __shared__ unsigned short lA[128 * 36];        // A tile bf16
    __shared__ unsigned short lBt[64 * 40];        // B^T tile bf16, 16B frags
    __shared__ unsigned short sStage[8 * 16 * 68]; // epilogue stage (Q/K path)

    const int tid  = threadIdx.x;
    const int wid  = tid >> 5;
    const int lane = tid & 31;
    const int hi   = lane >> 4;
    const int lm   = lane & 15;
    const int m0   = blockIdx.y * 128;             // token tile base
    const int n0   = blockIdx.x * 64;              // out-feature base (= head*64)

    v8f acc[4] = {};

    for (int kt = 0; kt < 32; ++kt) {
        const int k0 = kt * 32;
        // stage A: 128x32 fp32, float4 loads -> bf16 v4u stores
        #pragma unroll
        for (int i = 0; i < 4; ++i) {
            int e = tid + i * 256;                 // 1024 float4 chunks
            int r = e >> 3, c4 = (e & 7) * 4;
            const float4 f = *(const float4*)&X[(size_t)(m0 + r) * DM + k0 + c4];
            v4u pk = { f2bf(f.x), f2bf(f.y), f2bf(f.z), f2bf(f.w) };
            *(v4u*)&lA[r * 36 + c4] = pk;
        }
        // stage B transposed: thread owns a column of 4 K-rows -> one b64 store
        #pragma unroll
        for (int i = 0; i < 2; ++i) {
            int e = tid + i * 256;                 // 512 column chunks
            int c = e & 63, r4 = (e >> 6) * 4;
            v4u pk = { f2bf(W[(size_t)(k0 + r4 + 0) * DM + n0 + c]),
                       f2bf(W[(size_t)(k0 + r4 + 1) * DM + n0 + c]),
                       f2bf(W[(size_t)(k0 + r4 + 2) * DM + n0 + c]),
                       f2bf(W[(size_t)(k0 + r4 + 3) * DM + n0 + c]) };
            *(v4u*)&lBt[c * 40 + r4] = pk;
        }
        __syncthreads();

        // A fragment (lane row = lm within the wave's 16 rows)
        v16u au;
        const int arow = (wid * 16 + lm) * 36;
        #pragma unroll
        for (int p = 0; p < 8; ++p) {
            int kk = a_kidx(p, hi);
            au[2 * p]     = lA[arow + kk];
            au[2 * p + 1] = lA[arow + kk + 1];
        }
        #pragma unroll
        for (int nt = 0; nt < 4; ++nt) {
            v16u bu = ld16_lds(&lBt[(nt * 16 + lm) * 40 + hi * 16]);
            acc[nt] = wmma_bf16(au, bu, acc[nt]);
        }
        __syncthreads();
    }

    const int h = blockIdx.x;                      // head index (n0/64)
    if (z == 2) {
        // V: transposed [B,H,64,S]; 8 consecutive s per lane -> one 16B store
        const int token0 = m0 + wid * 16 + hi * 8;
        const int b = token0 >> 11, s = token0 & (SS - 1);
        #pragma unroll
        for (int nt = 0; nt < 4; ++nt) {
            const int d = nt * 16 + lm;
            const float bias = Bi[n0 + d];
            v8u pk;
            #pragma unroll
            for (int r = 0; r < 8; ++r) pk[r] = f2bf(acc[nt][r] + bias);
            *(v8u*)&Vt[(((size_t)b * HH + h) * DH + d) * SS + s] = pk;
        }
    } else {
        // Q/K: stage wave tile to LDS, then coalesced b32 row writes
        unsigned short* dst = (z == 0) ? Qh : Kh;
        #pragma unroll
        for (int nt = 0; nt < 4; ++nt) {
            const int d = nt * 16 + lm;
            const float bias = Bi[n0 + d];
            #pragma unroll
            for (int r = 0; r < 8; ++r)
                sStage[(wid * 16 + hi * 8 + r) * 68 + d] = f2bf(acc[nt][r] + bias);
        }
        #pragma unroll
        for (int rloc = 0; rloc < 16; ++rloc) {
            const int token = m0 + wid * 16 + rloc;
            const int b = token >> 11, s = token & (SS - 1);
            const unsigned int pair =
                *(const unsigned int*)&sStage[(wid * 16 + rloc) * 68 + lane * 2];
            *(unsigned int*)&dst[(((size_t)b * HH + h) * SS + s) * DH + lane * 2] = pair;
        }
    }
}

// =====================================================================
// Kernel 2: per (b,h, 16-row q tile): logits -> softmax -> atten out (NT),
// then ctx = P @ V via split-K WMMA + LDS ds_add_f32 reduction.
// =====================================================================
__global__ __launch_bounds__(256) void attn_kernel(
    const unsigned short* __restrict__ Qh, const unsigned short* __restrict__ Kh,
    const unsigned short* __restrict__ Vt,
    float* __restrict__ atten, unsigned short* __restrict__ ctx)
{
    __shared__ float          sP[16 * 2048];       // logits -> probs (128KB)
    __shared__ unsigned short sQ[16 * 64];         // q tile, bf16
    __shared__ float          sCtx[16 * 64];       // ctx reduction buffer

    const int tid  = threadIdx.x;
    const int wid  = tid >> 5;
    const int lane = tid & 31;
    const int hi   = lane >> 4;
    const int lm   = lane & 15;

    const int bid   = blockIdx.x;                  // B*H*(S/16) = 4096 blocks
    const int qt    = bid & 127;
    const int h     = (bid >> 7) & 15;
    const int b     = bid >> 11;
    const int qbase = qt * 16;
    const size_t bh = (size_t)b * HH + h;

    // async-copy q tile straight into LDS (128 x 16B chunks), zero sCtx
    if (tid < 128) {
        int r = tid >> 3, c8 = (tid & 7) * 8;
        async_copy_b128(&sQ[r * 64 + c8],
                        &Qh[(bh * SS + qbase + r) * DH + c8]);
    }
    sCtx[tid]       = 0.0f;
    sCtx[tid + 256] = 0.0f;
    sCtx[tid + 512] = 0.0f;
    sCtx[tid + 768] = 0.0f;
    wait_async0();
    __syncthreads();

    // two A fragments of Q (d = 0..31, 32..63)
    v16u aq[2];
    #pragma unroll
    for (int ks = 0; ks < 2; ++ks) {
        #pragma unroll
        for (int p = 0; p < 8; ++p) {
            int kk = ks * 32 + a_kidx(p, hi);
            aq[ks][2 * p]     = sQ[lm * 64 + kk];
            aq[ks][2 * p + 1] = sQ[lm * 64 + kk + 1];
        }
    }

    const float scale = 0.125f;                    // 1/sqrt(64)

    // ---- QK^T: wave owns 256 key columns = 16 tiles of 16x16 ----
    for (int t = 0; t < 16; ++t) {
        const int n0 = wid * 256 + t * 16;
        if (n0 > qbase + 15) {                     // fully masked tile: skip WMMA
            #pragma unroll
            for (int r = 0; r < 8; ++r)
                sP[(r + 8 * hi) * 2048 + n0 + lm] = -1.0e9f;
            continue;
        }
        if (t < 15)                                // prefetch next K tile row
            __builtin_prefetch(&Kh[(bh * SS + n0 + 16 + lm) * DH], 0, 1);
        v8f acc = {};
        #pragma unroll
        for (int ks = 0; ks < 2; ++ks) {
            const unsigned short* kp =
                &Kh[(bh * SS + n0 + lm) * DH + ks * 32 + hi * 16];
            v16u bu = *(const v16u*)kp;            // 32B-aligned global load
            acc = wmma_bf16(aq[ks], bu, acc);
        }
        #pragma unroll
        for (int r = 0; r < 8; ++r) {
            const int row = qbase + r + 8 * hi;
            const int col = n0 + lm;
            float lv = acc[r] * scale;
            if (col > row) lv -= 1.0e9f;           // causal (ref adds -1e9)
            sP[(r + 8 * hi) * 2048 + col] = lv;
        }
    }
    __syncthreads();

    // ---- softmax: wave handles 2 rows, 2 cols per lane per step ----
    #pragma unroll
    for (int rr = 0; rr < 2; ++rr) {
        const int rloc = wid * 2 + rr;
        const int row  = qbase + rloc;
        float m = -3.0e38f;
        for (int c = lane * 2; c < 2048; c += 64) {
            v2f lv = *(const v2f*)&sP[rloc * 2048 + c];
            m = fmaxf(m, fmaxf(lv.x, lv.y));
        }
        #pragma unroll
        for (int o = 16; o > 0; o >>= 1) m = fmaxf(m, __shfl_xor(m, o, 32));
        float sum = 0.0f;
        for (int c = lane * 2; c < 2048; c += 64) {
            v2f lv = *(const v2f*)&sP[rloc * 2048 + c];
            v2f ev = { __expf(lv.x - m), __expf(lv.y - m) };
            *(v2f*)&sP[rloc * 2048 + c] = ev;
            sum += ev.x + ev.y;
        }
        #pragma unroll
        for (int o = 16; o > 0; o >>= 1) sum += __shfl_xor(sum, o, 32);
        const float inv = 1.0f / sum;
        float* arow = &atten[(bh * SS + row) * SS];
        for (int c = lane * 2; c < 2048; c += 64) {
            v2f ev = *(const v2f*)&sP[rloc * 2048 + c];
            v2f pv = { ev.x * inv, ev.y * inv };
            *(v2f*)&sP[rloc * 2048 + c] = pv;
            __builtin_nontemporal_store(pv, (v2f*)&arow[c]);  // 536MB stream
        }
    }
    __syncthreads();

    // ---- ctx = P @ V, split-K across waves (wave owns its 256 keys) ----
    v8f accT[4] = {};
    for (int kt = 0; kt < 8; ++kt) {
        const int keybase = wid * 256 + kt * 32;
        v16u ap;
        #pragma unroll
        for (int p = 0; p < 8; ++p) {
            int kk = a_kidx(p, hi);
            ap[2 * p]     = f2bf(sP[lm * 2048 + keybase + kk]);
            ap[2 * p + 1] = f2bf(sP[lm * 2048 + keybase + kk + 1]);
        }
        #pragma unroll
        for (int nt = 0; nt < 4; ++nt) {
            const int d = nt * 16 + lm;
            const unsigned short* vp =
                &Vt[(bh * DH + d) * SS + keybase + hi * 16];  // contiguous keys
            v16u bu = *(const v16u*)vp;
            accT[nt] = wmma_bf16(ap, bu, accT[nt]);
        }
    }
    #pragma unroll
    for (int nt = 0; nt < 4; ++nt)
        #pragma unroll
        for (int r = 0; r < 8; ++r)
            atomicAdd(&sCtx[(r + 8 * hi) * 64 + nt * 16 + lm], accT[nt][r]); // ds_add_f32
    __syncthreads();

    // write ctx bf16 [B,S,1024], col = h*64+d (concat-heads), v4u stores
    {
        int e = tid * 4;
        int r = e >> 6, d = e & 63;
        v4u pk = { f2bf(sCtx[e]), f2bf(sCtx[e + 1]),
                   f2bf(sCtx[e + 2]), f2bf(sCtx[e + 3]) };
        *(v4u*)&ctx[((size_t)(b * SS + qbase + r)) * DM + h * DH + d] = pk;
    }
}

// =====================================================================
// Kernel 3: out = ctx @ wo + bo  (ctx already bf16, async-staged), fp32 NT out.
// =====================================================================
__global__ __launch_bounds__(256) void outproj_kernel(
    const unsigned short* __restrict__ ctx, const float* __restrict__ wo,
    const float* __restrict__ bo, float* __restrict__ out)
{
    __shared__ unsigned short lA[128 * 40];        // 16B-aligned async chunks
    __shared__ unsigned short lBt[64 * 40];
    __shared__ float          sOut[8 * 16 * 68];

    const int tid  = threadIdx.x;
    const int wid  = tid >> 5;
    const int lane = tid & 31;
    const int hi   = lane >> 4;
    const int lm   = lane & 15;
    const int m0   = blockIdx.y * 128;
    const int n0   = blockIdx.x * 64;

    v8f acc[4] = {};

    for (int kt = 0; kt < 32; ++kt) {
        const int k0 = kt * 32;
        // async stage A: 512 x 16B LDS chunks direct from global bf16 ctx
        #pragma unroll
        for (int i = 0; i < 2; ++i) {
            int e = tid + i * 256;
            int r = e >> 2, c8 = (e & 3) * 8;
            async_copy_b128(&lA[r * 40 + c8],
                            &ctx[(size_t)(m0 + r) * DM + k0 + c8]);
        }
        // stage B^T: column of 4 K-rows per thread -> one b64 store
        #pragma unroll
        for (int i = 0; i < 2; ++i) {
            int e = tid + i * 256;
            int c = e & 63, r4 = (e >> 6) * 4;
            v4u pk = { f2bf(wo[(size_t)(k0 + r4 + 0) * DM + n0 + c]),
                       f2bf(wo[(size_t)(k0 + r4 + 1) * DM + n0 + c]),
                       f2bf(wo[(size_t)(k0 + r4 + 2) * DM + n0 + c]),
                       f2bf(wo[(size_t)(k0 + r4 + 3) * DM + n0 + c]) };
            *(v4u*)&lBt[c * 40 + r4] = pk;
        }
        wait_async0();
        __syncthreads();

        v16u au;
        const int arow = (wid * 16 + lm) * 40;
        #pragma unroll
        for (int p = 0; p < 8; ++p) {
            int kk = a_kidx(p, hi);
            au[2 * p]     = lA[arow + kk];
            au[2 * p + 1] = lA[arow + kk + 1];
        }
        #pragma unroll
        for (int nt = 0; nt < 4; ++nt) {
            v16u bu = ld16_lds(&lBt[(nt * 16 + lm) * 40 + hi * 16]);
            acc[nt] = wmma_bf16(au, bu, acc[nt]);
        }
        __syncthreads();
    }

    // stage f32 tile per wave, then coalesced nontemporal float2 row writes
    #pragma unroll
    for (int nt = 0; nt < 4; ++nt) {
        const float bias = bo[n0 + nt * 16 + lm];
        #pragma unroll
        for (int r = 0; r < 8; ++r)
            sOut[(wid * 16 + hi * 8 + r) * 68 + nt * 16 + lm] = acc[nt][r] + bias;
    }
    #pragma unroll
    for (int rloc = 0; rloc < 16; ++rloc) {
        const int token = m0 + wid * 16 + rloc;
        v2f val = *(const v2f*)&sOut[(wid * 16 + rloc) * 68 + lane * 2];
        __builtin_nontemporal_store(val, (v2f*)&out[(size_t)token * DM + n0 + lane * 2]);
    }
}

// =====================================================================
extern "C" void kernel_launch(void* const* d_in, const int* in_sizes, int n_in,
                              void* d_out, int out_size, void* d_ws, size_t ws_size,
                              hipStream_t stream) {
    (void)in_sizes; (void)n_in; (void)out_size; (void)ws_size;
    const float* q   = (const float*)d_in[0];
    const float* k   = (const float*)d_in[1];
    const float* v   = (const float*)d_in[2];
    // d_in[3] = mask: causal, computed analytically in-kernel
    const float* wq  = (const float*)d_in[4];
    const float* bq  = (const float*)d_in[5];
    const float* wk  = (const float*)d_in[6];
    const float* bk  = (const float*)d_in[7];
    const float* wv_ = (const float*)d_in[8];
    const float* bv  = (const float*)d_in[9];
    const float* wo  = (const float*)d_in[10];
    const float* bo  = (const float*)d_in[11];

    float* out   = (float*)d_out;
    float* atten = out + (size_t)BB * SS * DM;        // out first, then atten

    const size_t E = (size_t)BB * SS * DM;            // 4,194,304 elems per buffer
    unsigned short* wsu = (unsigned short*)d_ws;      // 4 x 8MB bf16 buffers
    unsigned short* Qh  = wsu;
    unsigned short* Kh  = wsu + E;
    unsigned short* Vt  = wsu + 2 * E;
    unsigned short* ctx = wsu + 3 * E;

    proj_kernel<<<dim3(16, 32, 3), 256, 0, stream>>>(q, k, v, wq, bq, wk, bk,
                                                     wv_, bv, Qh, Kh, Vt);
    attn_kernel<<<dim3(BB * HH * (SS / 16)), 256, 0, stream>>>(Qh, Kh, Vt, atten, ctx);
    outproj_kernel<<<dim3(16, 32), 256, 0, stream>>>(ctx, wo, bo, out);
}
